// MultiheadAttention_18760417149600
// MI455X (gfx1250) — matrix operand order repaired and checked by reading, not compiled, not run on hardware
//
#include <hip/hip_runtime.h>
#include <hip/hip_bf16.h>
#include <cstdint>

// ---------------------------------------------------------------------------
// Types
// ---------------------------------------------------------------------------
typedef __bf16 bf16;
typedef __attribute__((ext_vector_type(16))) __bf16        v16bf;
typedef __attribute__((ext_vector_type(8)))  float         v8f;
typedef __attribute__((ext_vector_type(4)))  unsigned int  u32x4;
typedef __attribute__((ext_vector_type(2)))  unsigned int  u32x2;
typedef __attribute__((ext_vector_type(4)))  float         f32x4;

union Frag {
  v16bf v;
  u32x4 q[2];
};

__device__ __forceinline__ v8f wmma_bf16(v16bf a, v16bf b, v8f c) {
  // D = A(16x32 bf16) * B(32x16 bf16) + C(16x16 f32)
  return __builtin_amdgcn_wmma_f32_16x16x32_bf16(
      /*neg_a=*/false, a, /*neg_b=*/false, b,
      /*c_mod=*/(short)0, c, /*reuse_a=*/false, /*reuse_b=*/false);
}

// Intra-wave LDS store->load ordering (DS ops are in-order per wave).
__device__ __forceinline__ void wave_ds_fence() {
  __builtin_amdgcn_wave_barrier();
  asm volatile("s_wait_dscnt 0" ::: "memory");
  __builtin_amdgcn_wave_barrier();
}

// Async global->LDS copy (ASYNCcnt tracked, bypasses VGPRs). Per-lane 16B.
// ISA 15.18.3: GLOBAL_LOAD_ASYNC_TO_LDS_B128, GV mode (64-bit VGPR address).
__device__ __forceinline__ void async_copy_b128(unsigned lds_addr,
                                                const void* gaddr) {
  asm volatile("global_load_async_to_lds_b128 %0, %1, off"
               :: "v"(lds_addr), "v"(gaddr)
               : "memory");
}
__device__ __forceinline__ void wait_async0() {
  asm volatile("s_wait_asynccnt 0" ::: "memory");
}

// ---------------------------------------------------------------------------
// 16-lane-half reductions via v_permlane16_b32 (VALU) instead of ds_bpermute.
// Selector nibbles: lane i reads lane sel[i] of its 16-lane group.
// ---------------------------------------------------------------------------
#if __has_builtin(__builtin_amdgcn_permlane16)
__device__ __forceinline__ float perm16f(float x, unsigned lo, unsigned hi) {
  unsigned u = __float_as_uint(x);
  u = __builtin_amdgcn_permlane16(u, u, lo, hi, false, false);
  return __uint_as_float(u);
}
__device__ __forceinline__ float red16_max(float x) {
  x = fmaxf(x, perm16f(x, 0x67452301u, 0xefcdab89u)); // xor 1
  x = fmaxf(x, perm16f(x, 0x54761032u, 0xdcfe98bau)); // xor 2
  x = fmaxf(x, perm16f(x, 0x32107654u, 0xba98fedcu)); // xor 4
  x = fmaxf(x, perm16f(x, 0xfedcba98u, 0x76543210u)); // xor 8
  return x;
}
__device__ __forceinline__ float red16_sum(float x) {
  x += perm16f(x, 0x67452301u, 0xefcdab89u);
  x += perm16f(x, 0x54761032u, 0xdcfe98bau);
  x += perm16f(x, 0x32107654u, 0xba98fedcu);
  x += perm16f(x, 0xfedcba98u, 0x76543210u);
  return x;
}
#else
__device__ __forceinline__ float red16_max(float x) {
  for (int d = 8; d >= 1; d >>= 1) x = fmaxf(x, __shfl_xor(x, d, 32));
  return x;
}
__device__ __forceinline__ float red16_sum(float x) {
  for (int d = 8; d >= 1; d >>= 1) x += __shfl_xor(x, d, 32);
  return x;
}
#endif

// ---------------------------------------------------------------------------
// fp32 -> bf16 conversion (vectorized)
// ---------------------------------------------------------------------------
__global__ __launch_bounds__(256)
void cvt_f32_bf16(const float* __restrict__ src, bf16* __restrict__ dst, int n) {
  int i = (blockIdx.x * 256 + threadIdx.x) * 4;
  if (i >= n) return;
  f32x4 f = *(const f32x4*)(src + i);
  union { bf16 e[4]; u32x2 u; } o;
  o.e[0] = (bf16)f[0];
  o.e[1] = (bf16)f[1];
  o.e[2] = (bf16)f[2];
  o.e[3] = (bf16)f[3];
  *(u32x2*)(dst + i) = o.u;
}

// ---------------------------------------------------------------------------
// GEMM: C[M,N] = A[M,K] @ W[N,K]^T + bias[N]
// 128x128 block tile, 256 threads = 8 waves as 4(m) x 2(n).
// Each wave: 32x64 sub-tile => 8 v_wmma per 32-wide K step.
// Double-buffered LDS staging via global_load_async_to_lds_b128.
// ---------------------------------------------------------------------------
template <bool OUT_BF16>
__global__ __launch_bounds__(256)
void gemm_bf16_wmma(const bf16* __restrict__ A, const bf16* __restrict__ W,
                    const float* __restrict__ bias, void* __restrict__ Cout,
                    int M, int N, int K) {
  __shared__ bf16 Ash[2][128][32];
  __shared__ bf16 Bsh[2][128][32];

  const int tid  = threadIdx.x;
  const int lane = tid & 31;
  const int w    = tid >> 5;
  const int wm   = w & 3;             // wave row index   (4)
  const int wn   = w >> 2;            // wave col index   (2)
  const int bm   = blockIdx.y * 128;
  const int bn   = blockIdx.x * 128;
  const int m    = lane & 15;
  const int half = lane >> 4;

  v8f acc[2][4];
  {
    v8f z = {};
#pragma unroll
    for (int mi = 0; mi < 2; ++mi)
#pragma unroll
      for (int j = 0; j < 4; ++j) acc[mi][j] = z;
  }

  const int sr = tid >> 1;            // staging row 0..127
  const int ss = (tid & 1) * 16;      // staging col segment (16 elems)
  const bf16* aptr = A + (size_t)(bm + sr) * K + ss;
  const bf16* wptr = W + (size_t)(bn + sr) * K + ss;

  auto stage = [&](int buf, int k0) {
    async_copy_b128((unsigned)(uintptr_t)&Ash[buf][sr][ss],     aptr + k0);
    async_copy_b128((unsigned)(uintptr_t)&Ash[buf][sr][ss + 8], aptr + k0 + 8);
    async_copy_b128((unsigned)(uintptr_t)&Bsh[buf][sr][ss],     wptr + k0);
    async_copy_b128((unsigned)(uintptr_t)&Bsh[buf][sr][ss + 8], wptr + k0 + 8);
  };

  stage(0, 0);
  wait_async0();
  __syncthreads();

  int buf = 0;
  for (int k0 = 0; k0 < K; k0 += 32, buf ^= 1) {
    if (k0 + 32 < K) stage(buf ^ 1, k0 + 32);  // overlap copy with compute

    // A fragments: 16x32 layout; lanes 0-15 K chunks [0..7],[16..23],
    // lanes 16-31 K chunks [8..15],[24..31].
    Frag a[2];
    const int kb = half * 8;
#pragma unroll
    for (int mi = 0; mi < 2; ++mi) {
      a[mi].q[0] = *(const u32x4*)&Ash[buf][wm * 32 + mi * 16 + m][kb];
      a[mi].q[1] = *(const u32x4*)&Ash[buf][wm * 32 + mi * 16 + m][kb + 16];
    }
    // B fragments: 32x16 layout; lane holds column n=lane&15, K rows half*8..+7, 16+half*8..+7.
    const int kb2 = half * 8;
#pragma unroll
    for (int j = 0; j < 4; ++j) {
      Frag bfr;
      bfr.q[0] = *(const u32x4*)&Bsh[buf][wn * 64 + j * 16 + m][kb2];
      bfr.q[1] = *(const u32x4*)&Bsh[buf][wn * 64 + j * 16 + m][kb2 + 16];
#pragma unroll
      for (int mi = 0; mi < 2; ++mi)
        acc[mi][j] = wmma_bf16(a[mi].v, bfr.v, acc[mi][j]);
    }
    wait_async0();
    __syncthreads();
  }

  // Epilogue: C layout => lane holds column m (per 16-tile), rows r + 8*half.
#pragma unroll
  for (int mi = 0; mi < 2; ++mi)
#pragma unroll
    for (int j = 0; j < 4; ++j) {
      const int n = bn + wn * 64 + j * 16 + m;
      const float bz = bias[n];
#pragma unroll
      for (int r = 0; r < 8; ++r) {
        const int row = bm + wm * 32 + mi * 16 + half * 8 + r;
        const float val = acc[mi][j][r] + bz;
        if (OUT_BF16)
          ((bf16*)Cout)[(size_t)row * N + n] = (bf16)val;
        else
          ((float*)Cout)[(size_t)row * N + n] = val;
      }
    }
}

// ---------------------------------------------------------------------------
// RoPE in place on q (B,T,32,64) and k (B,T,8,64), bf16.
// One 32-lane wave per head-row; lane i handles the (i, i+32) pair.
// ---------------------------------------------------------------------------
__global__ __launch_bounds__(128)
void rope_kernel(bf16* __restrict__ qb, bf16* __restrict__ kb, int T) {
  const int unit = blockIdx.x * 4 + (threadIdx.x >> 5);
  const int lane = threadIdx.x & 31;
  const int b   = unit / (T * 40);
  const int rem = unit - b * (T * 40);
  const int t   = rem / 40;
  const int hh  = rem - t * 40;

  bf16* p = (hh < 32)
      ? qb + (((size_t)b * T + t) * 32 + hh) * 64
      : kb + (((size_t)b * T + t) * 8 + (hh - 32)) * 64;

  // inv_freq = 10000^(-lane/32) = exp(-lane * ln(10000)/32)
  const float ang = (float)t * __expf((float)lane * (-9.2103403719761836f / 32.0f));
  float sn, cs;
  __sincosf(ang, &sn, &cs);
  const float x1 = (float)p[lane];
  const float x2 = (float)p[lane + 32];
  p[lane]      = (bf16)(x1 * cs - x2 * sn);
  p[lane + 32] = (bf16)(x2 * cs + x1 * sn);
}

// ---------------------------------------------------------------------------
// Flash attention (causal, GQA 4:1). Grid (T/64, H, B), 128 threads = 4 waves.
// Wave w owns query rows [qblk+16w, +16) x head-dim 64.
// Per 32-key step: 4 WMMA for scores (K-dim 64) + 4 WMMA for P*V.
// ---------------------------------------------------------------------------
__global__ __launch_bounds__(128)
void flash_attn_wmma(const bf16* __restrict__ Q, const bf16* __restrict__ Kc,
                     const bf16* __restrict__ Vc, bf16* __restrict__ O, int T) {
  constexpr int H = 32, KVH = 8, HD = 64;
  __shared__ bf16 Ksh[32][64];    // [key][d]
  __shared__ bf16 Vtsh[64][32];   // [d][key] (transposed for B-layout reads)
  __shared__ bf16 Psh[4][16][32]; // per-wave C-layout -> A-layout transpose

  const int tid  = threadIdx.x;
  const int lane = tid & 31;
  const int w    = tid >> 5;
  const int m    = lane & 15;
  const int half = lane >> 4;
  const int qblk = blockIdx.x * 64;
  const int h    = blockIdx.y;
  const int b    = blockIdx.z;
  const int kvh  = h >> 2;           // H/KVH = 4
  const int qrow = qblk + w * 16;

  // Q fragments (A-layout): frag0 = d 0..31, frag1 = d 32..63.
  Frag qf0, qf1;
  {
    const size_t qoff = (((size_t)b * T + (qrow + m)) * H + h) * HD;
    const u32x4* qp = (const u32x4*)(Q + qoff);
    qf0.q[0] = qp[half];       // d = half*8 .. +7
    qf0.q[1] = qp[half + 2];   // d = 16 + half*8
    qf1.q[0] = qp[half + 4];   // d = 32 + half*8
    qf1.q[1] = qp[half + 6];   // d = 48 + half*8
  }

  float mstat[8], lstat[8];
  v8f accO[4];
  {
    v8f z = {};
#pragma unroll
    for (int j = 0; j < 4; ++j) accO[j] = z;
#pragma unroll
    for (int r = 0; r < 8; ++r) { mstat[r] = -3.0e38f; lstat[r] = 0.0f; }
  }

  const int skey = tid >> 2;         // staging: key 0..31
  const int sseg = (tid & 3) * 16;   // staging: 16-elem d segment
  const size_t kvb = (size_t)b * T * (KVH * HD) + (size_t)kvh * HD;

  const int kend = qblk + 64;
  for (int kt = 0; kt < kend; kt += 32) {
    // Stage K tile asynchronously; V tile transposed through registers.
    {
      const size_t roff = kvb + (size_t)(kt + skey) * (KVH * HD) + sseg;
      async_copy_b128((unsigned)(uintptr_t)&Ksh[skey][sseg],     Kc + roff);
      async_copy_b128((unsigned)(uintptr_t)&Ksh[skey][sseg + 8], Kc + roff + 8);
      union { u32x4 u[2]; bf16 e[16]; } vr;
      const u32x4* vs = (const u32x4*)(Vc + roff);
      vr.u[0] = vs[0];
      vr.u[1] = vs[1];
#pragma unroll
      for (int e = 0; e < 16; ++e) Vtsh[sseg + e][skey] = vr.e[e];
      if (kt + 32 < kend) {
        __builtin_prefetch(Kc + roff + 32 * (KVH * HD), 0, 3);
        __builtin_prefetch(Vc + roff + 32 * (KVH * HD), 0, 3);
      }
    }
    wait_async0();
    __syncthreads();

    if (kt <= qrow + 15) {  // tile intersects this wave's causal region
      // Scores: S = Q * K^T (two 16x16 key sub-tiles, K-dim 64 = 2 WMMA each)
      v8f s[2];
      const int db = half * 8;
#pragma unroll
      for (int jt = 0; jt < 2; ++jt) {
        Frag kb0, kb1;
        const int key = jt * 16 + m;
        kb0.q[0] = *(const u32x4*)&Ksh[key][db];
        kb0.q[1] = *(const u32x4*)&Ksh[key][db + 16];
        kb1.q[0] = *(const u32x4*)&Ksh[key][db + 32];
        kb1.q[1] = *(const u32x4*)&Ksh[key][db + 48];
        v8f z = {};
        z = wmma_bf16(qf0.v, kb0.v, z);
        z = wmma_bf16(qf1.v, kb1.v, z);
        s[jt] = z;
      }
      // Causal mask + 1/sqrt(64) scale.
#pragma unroll
      for (int jt = 0; jt < 2; ++jt) {
        const int keyg = kt + jt * 16 + m;
#pragma unroll
        for (int r = 0; r < 8; ++r) {
          const int rowg = qrow + half * 8 + r;
          const float val = s[jt][r] * 0.125f;
          s[jt][r] = (keyg <= rowg) ? val : -3.0e38f;
        }
      }
      // Online softmax: row reductions via v_permlane16 (VALU, no DS traffic).
#pragma unroll
      for (int r = 0; r < 8; ++r) {
        const float t = red16_max(fmaxf(s[0][r], s[1][r]));
        const float mnew = fmaxf(mstat[r], t);
        const float sc = __expf(mstat[r] - mnew);
        mstat[r] = mnew;
        const float p0 = __expf(s[0][r] - mnew);
        const float p1 = __expf(s[1][r] - mnew);
        s[0][r] = p0;
        s[1][r] = p1;
        lstat[r] = lstat[r] * sc + red16_sum(p0 + p1);
#pragma unroll
        for (int j = 0; j < 4; ++j) accO[j][r] *= sc;
      }
      // P: C-layout -> A-layout via per-wave LDS buffer.
#pragma unroll
      for (int jt = 0; jt < 2; ++jt)
#pragma unroll
        for (int r = 0; r < 8; ++r)
          Psh[w][half * 8 + r][jt * 16 + m] = (bf16)s[jt][r];
      wave_ds_fence();
      Frag pa;
      pa.q[0] = *(const u32x4*)&Psh[w][m][half * 8];
      pa.q[1] = *(const u32x4*)&Psh[w][m][half * 8 + 16];
      // accO += P(16x32) * V(32x64), accumulating through the WMMA C operand.
#pragma unroll
      for (int jn = 0; jn < 4; ++jn) {
        Frag vb;
        vb.q[0] = *(const u32x4*)&Vtsh[jn * 16 + m][half * 8];
        vb.q[1] = *(const u32x4*)&Vtsh[jn * 16 + m][half * 8 + 16];
        accO[jn] = wmma_bf16(pa.v, vb.v, accO[jn]);
      }
    }
    __syncthreads();
  }

  // Normalize and store attn output (B,T,H*HD) in bf16 for the O-projection.
#pragma unroll
  for (int jn = 0; jn < 4; ++jn)
#pragma unroll
    for (int r = 0; r < 8; ++r) {
      const int rowg = qrow + half * 8 + r;
      const size_t off =
          ((size_t)b * T + rowg) * (H * HD) + (size_t)h * HD + jn * 16 + m;
      O[off] = (bf16)(accO[jn][r] / lstat[r]);
    }
}

// ---------------------------------------------------------------------------
// Host launch
// ---------------------------------------------------------------------------
extern "C" void kernel_launch(void* const* d_in, const int* in_sizes, int n_in,
                              void* d_out, int out_size, void* d_ws, size_t ws_size,
                              hipStream_t stream) {
  constexpr int B = 2, T = 2048, D = 2048, H = 32, KVH = 8, HD = 64;
  constexpr int KVD = KVH * HD;         // 512
  constexpr size_t M = (size_t)B * T;   // 4096

  const float* x  = (const float*)d_in[0];
  const float* Wq = (const float*)d_in[1];
  const float* bq = (const float*)d_in[2];
  const float* Wk = (const float*)d_in[3];
  const float* bk = (const float*)d_in[4];
  const float* Wv = (const float*)d_in[5];
  const float* bv = (const float*)d_in[6];
  const float* Wo = (const float*)d_in[7];
  const float* bo = (const float*)d_in[8];
  float* out = (float*)d_out;
  (void)in_sizes; (void)n_in; (void)out_size; (void)ws_size;

  // Workspace carve (~76 MB total)
  char* ws = (char*)d_ws;
  size_t off = 0;
  auto carve = [&](size_t bytes) { char* p = ws + off; off += bytes; return p; };
  bf16* x_bf  = (bf16*)carve(M * D * 2);
  bf16* wq_bf = (bf16*)carve((size_t)D * D * 2);
  bf16* wk_bf = (bf16*)carve((size_t)KVD * D * 2);
  bf16* wv_bf = (bf16*)carve((size_t)KVD * D * 2);
  bf16* wo_bf = (bf16*)carve((size_t)D * D * 2);
  bf16* q_bf  = (bf16*)carve(M * D * 2);
  bf16* k_bf  = (bf16*)carve(M * KVD * 2);
  bf16* v_bf  = (bf16*)carve(M * KVD * 2);
  bf16* a_bf  = (bf16*)carve(M * D * 2);

  auto cvt = [&](const float* s, bf16* d, size_t n) {
    cvt_f32_bf16<<<dim3((unsigned)(n / 1024)), 256, 0, stream>>>(s, d, (int)n);
  };
  cvt(x,  x_bf,  M * D);
  cvt(Wq, wq_bf, (size_t)D * D);
  cvt(Wk, wk_bf, (size_t)KVD * D);
  cvt(Wv, wv_bf, (size_t)KVD * D);
  cvt(Wo, wo_bf, (size_t)D * D);

  // QKV projections (bf16 out, fused bias). 128x128 tiles.
  gemm_bf16_wmma<true><<<dim3(D / 128, M / 128), 256, 0, stream>>>(
      x_bf, wq_bf, bq, q_bf, (int)M, D, D);
  gemm_bf16_wmma<true><<<dim3(KVD / 128, M / 128), 256, 0, stream>>>(
      x_bf, wk_bf, bk, k_bf, (int)M, KVD, D);
  gemm_bf16_wmma<true><<<dim3(KVD / 128, M / 128), 256, 0, stream>>>(
      x_bf, wv_bf, bv, v_bf, (int)M, KVD, D);

  // RoPE on q and k
  rope_kernel<<<(B * T * (H + KVH)) / 4, 128, 0, stream>>>(q_bf, k_bf, T);

  // Causal flash attention
  flash_attn_wmma<<<dim3(T / 64, H, B), 128, 0, stream>>>(q_bf, k_bf, v_bf, a_bf, T);

  // Output projection (fp32 out, fused bias)
  gemm_bf16_wmma<false><<<dim3(D / 128, M / 128), 256, 0, stream>>>(
      a_bf, wo_bf, bo, out, (int)M, D, D);
}
